// CrossAttention_35837207117938
// MI455X (gfx1250) — matrix operand order repaired and checked by reading, compiled, not run on hardware
//
#include <hip/hip_runtime.h>
#include <hip/hip_bf16.h>
#include <math.h>

// ---------------- problem constants ----------------
#define B_   2
#define NQ_  2048
#define NK_  2048
#define D_   1024
#define H_   16
#define HD_  64

#define LDS_P 40            // padded LDS row stride in halfs (32 data + 8 pad)
#define A_BUF (128 * LDS_P) // halfs per A LDS buffer (10240 B)
#define B_BUF (64 * LDS_P)  // halfs per B LDS buffer (5120 B)

typedef __attribute__((ext_vector_type(16))) __bf16 bf16x16;
typedef __attribute__((ext_vector_type(8)))  __bf16 bf16x8;
typedef __attribute__((ext_vector_type(8)))  float  f32x8;
typedef __attribute__((ext_vector_type(4)))  float  f32x4;
typedef __attribute__((ext_vector_type(4)))  unsigned int u32x4;
typedef __attribute__((ext_vector_type(8)))  int i32x8;
typedef __attribute__((ext_vector_type(4)))  int i32x4;

// ---------------------------------------------------------------------------
// TDM: async 2-D bf16 tile load (rows x 32 elems) global -> LDS.
// D# per CDNA5 ISA 8.3-8.5. pad_enable inserts 4 DWORDs of LDS padding every
// 16 DWORDs stored => LDS row stride = 32 halfs data + 8 halfs pad = LDS_P.
// ---------------------------------------------------------------------------
__device__ __forceinline__ void tdm_load_tile_bf16(
    unsigned lds_off, const void* gsrc, int rows, long row_stride_elems)
{
  unsigned long long ga = (unsigned long long)(size_t)gsrc;
  u32x4 g0;
  g0.x = 1u;                                               // count=1 (valid user D#)
  g0.y = lds_off;                                          // lds_addr (bytes)
  g0.z = (unsigned)(ga & 0xffffffffu);                     // global_addr[31:0]
  g0.w = (unsigned)((ga >> 32) & 0x01ffffffu) | (2u << 30);// addr[56:32] | type=2
  i32x8 g1;
  g1[0] = (1 << 16)            // data_size = 1 -> 2 bytes
        | (1 << 20)            // pad_enable
        | (3 << 22)            // pad_interval: 16 DWORDs (64B) between pads
        | (3 << 25);           // pad_amount:   4 DWORDs (16B) of pad
  g1[1] = (int)(32u << 16);                    // tensor_dim0 = 32 elems
  g1[2] = (int)((unsigned)rows << 16);         // tensor_dim1 = rows
  g1[3] = (int)(32u << 16);                    // tile_dim0 = 32 elems
  g1[4] = rows;                                // tile_dim1 = rows (tile_dim2=0)
  g1[5] = (int)(unsigned)row_stride_elems;     // tensor_dim0_stride lo
  g1[6] = (int)((unsigned)((unsigned long long)row_stride_elems >> 32) & 0xffff);
  g1[7] = 0;
  i32x4 z4 = {0, 0, 0, 0};
#if __clang_major__ >= 23
  i32x8 z8 = {};
  __builtin_amdgcn_tensor_load_to_lds(g0, g1, z4, z4, z8, 0);
#else
  __builtin_amdgcn_tensor_load_to_lds(g0, g1, z4, z4, 0);
#endif
}

// ---------------------------------------------------------------------------
// Block GEMM:  C[m,n] = scale * sum_k A[m,k] * Bt[n,k]  (+ bias[n])
// Block tile 128(M) x 64(N), K-step 32. 8 waves in a 4x2 grid; each wave owns
// a 2x2 set of 16x16 WMMA tiles (4 x v_wmma_f32_16x16x32_bf16 per K-step).
//   USE_TDM : A,Bt bf16; tiles staged by the Tensor Data Mover with ping-pong
//             LDS buffers — next tile pair issued before computing on the
//             current one; s_wait_tensorcnt(1) keeps one pair in flight.
//   else    : A is f32 (converted on load), Bt bf16, manual staging.
// ---------------------------------------------------------------------------
template<bool USE_TDM, bool OUT_BF16, bool HAS_BIAS>
__device__ __forceinline__ void gemm_block(
    const void* __restrict__ Ap, long lda,
    const __bf16* __restrict__ Bt, long ldb,
    void* __restrict__ Cp, long ldc,
    const float* __restrict__ bias, float scale, int K,
    long m0, long n0,
    __bf16* As, __bf16* Bs)
{
  const int tid  = threadIdx.x;
  const int lane = tid & 31;
  const int w    = tid >> 5;     // wave 0..7
  const int wm   = w >> 1;       // 0..3 : 32-row slab
  const int wn   = w & 1;        // 0..1 : 32-col slab

  const int r  = lane & 15;
  const int hs = lane >> 4;      // half-select

  f32x8 acc[2][2] = {};

  const unsigned as_off = (unsigned)(size_t)(void*)As;  // LDS byte offsets
  const unsigned bs_off = (unsigned)(size_t)(void*)Bs;

  // TDM prologue: prime buffer 0 with the first tile pair
  if (USE_TDM) {
    const __bf16* A = (const __bf16*)Ap;
    if (tid == 0)   // TDM ignores EXEC; other waves branch around
      tdm_load_tile_bf16(as_off, A + m0 * lda, 128, lda);
    if (tid == 32)
      tdm_load_tile_bf16(bs_off, Bt + n0 * ldb, 64, ldb);
  }

  for (int k0 = 0; k0 < K; k0 += 32) {
    const int cur = USE_TDM ? ((k0 >> 5) & 1) : 0;

    if (USE_TDM) {
      const __bf16* A = (const __bf16*)Ap;
      if (k0 + 32 < K) {
        const int nxt = cur ^ 1;
        if (tid == 0)
          tdm_load_tile_bf16(as_off + (unsigned)nxt * (A_BUF * 2),
                             A + m0 * lda + (k0 + 32), 128, lda);
        if (tid == 32)
          tdm_load_tile_bf16(bs_off + (unsigned)nxt * (B_BUF * 2),
                             Bt + n0 * ldb + (k0 + 32), 64, ldb);
        __builtin_amdgcn_s_wait_tensorcnt(1);  // current pair landed; next in flight
      } else {
        __builtin_amdgcn_s_wait_tensorcnt(0);
      }
    } else {
      // ---- stage A tile (128 x 32 f32 -> bf16) ----
      const float* A = (const float*)Ap;
#pragma unroll
      for (int c = 0; c < 2; ++c) {
        const int id  = tid + c * 256;       // 0..511
        const int row = id >> 2;             // 0..127
        const int col = (id & 3) << 3;       // 0,8,16,24
        const float* src = A + (m0 + row) * lda + (k0 + col);
        f32x4 f0 = *(const f32x4*)(src);
        f32x4 f1 = *(const f32x4*)(src + 4);
        bf16x8 h;
#pragma unroll
        for (int i = 0; i < 4; ++i) { h[i] = (__bf16)f0[i]; h[4 + i] = (__bf16)f1[i]; }
        *(bf16x8*)&As[row * LDS_P + col] = h;
        if (k0 + 32 < K) __builtin_prefetch(src + 32, 0, 1);
      }
      // ---- stage Bt tile (64 x 32 bf16) ----
      {
        const int row = tid >> 2;            // 0..63
        const int col = (tid & 3) << 3;
        *(bf16x8*)&Bs[row * LDS_P + col] =
            *(const bf16x8*)(Bt + (n0 + row) * ldb + (k0 + col));
      }
    }
    __syncthreads();

    const __bf16* Ac = As + cur * A_BUF;
    const __bf16* Bc = Bs + cur * B_BUF;

    // ---- gather fragments (ISA 7.12.2, wave32 16-bit layouts) ----
    bf16x16 af[2], bf[2];
#pragma unroll
    for (int i = 0; i < 2; ++i) {
      const __bf16* ar = &Ac[(wm * 32 + i * 16 + r) * LDS_P + (hs << 3)];
      bf16x8 a0 = *(const bf16x8*)(ar);
      bf16x8 a1 = *(const bf16x8*)(ar + 16);
#pragma unroll
      for (int e = 0; e < 8; ++e) { af[i][e] = a0[e]; af[i][8 + e] = a1[e]; }
    }
#pragma unroll
    for (int j = 0; j < 2; ++j) {
      const __bf16* br = &Bc[(wn * 32 + j * 16 + r) * LDS_P + (hs << 3)];
      bf16x8 b0 = *(const bf16x8*)(br);
      bf16x8 b1 = *(const bf16x8*)(br + 16);
#pragma unroll
      for (int e = 0; e < 8; ++e) { bf[j][e] = b0[e]; bf[j][8 + e] = b1[e]; }
    }

#pragma unroll
    for (int i = 0; i < 2; ++i)
#pragma unroll
      for (int j = 0; j < 2; ++j)
        acc[i][j] = __builtin_amdgcn_wmma_f32_16x16x32_bf16(
            false, af[i], false, bf[j], (short)0, acc[i][j], false, false);
    __syncthreads();  // protects buffer reuse (TDM) / LDS overwrite (manual)
  }

  // ---- epilogue: VGPR e -> M = base + hs*8 + e, N = base + (lane&15) ----
#pragma unroll
  for (int i = 0; i < 2; ++i) {
    const long mBase = m0 + wm * 32 + i * 16 + (hs << 3);
#pragma unroll
    for (int j = 0; j < 2; ++j) {
      const long n = n0 + wn * 32 + j * 16 + r;
      const float bb = HAS_BIAS ? bias[n] : 0.0f;
#pragma unroll
      for (int e = 0; e < 8; ++e) {
        float v = acc[i][j][e] * scale + bb;
        if (OUT_BF16) ((__bf16*)Cp)[(mBase + e) * ldc + n] = (__bf16)v;
        else          ((float*) Cp)[(mBase + e) * ldc + n] = v;
      }
    }
  }
}

// ---------------- GEMM wrappers ----------------

// projections: Y(M x 1024 f32) = X(f32) @ W^T + b   (manual staging, f32->bf16)
__global__ void xattn_proj_gemm(const float* __restrict__ X,
                                const __bf16* __restrict__ Wb,
                                const float* __restrict__ bias,
                                float* __restrict__ Y)
{
  __shared__ __bf16 As[A_BUF];
  __shared__ __bf16 Bs[B_BUF];
  gemm_block<false, false, true>(X, D_, Wb, D_, Y, D_, bias, 1.0f, D_,
                                 (long)blockIdx.y * 128, (long)blockIdx.x * 64, As, Bs);
}

// scores: S = (Q K^T) * HD^-0.5 per (b,h)  (TDM ping-pong staging)
__global__ void xattn_qk_gemm(const __bf16* __restrict__ Qb,
                              const __bf16* __restrict__ Kb,
                              float* __restrict__ S)
{
  __shared__ __bf16 As[2 * A_BUF];
  __shared__ __bf16 Bs[2 * B_BUF];
  const long bh = blockIdx.z;
  gemm_block<true, false, false>(Qb + bh * (long)NQ_ * HD_, HD_,
                                 Kb + bh * (long)NK_ * HD_, HD_,
                                 S + bh * (long)NQ_ * NK_, NK_,
                                 nullptr, 0.125f, HD_,
                                 (long)blockIdx.y * 128, (long)blockIdx.x * 64, As, Bs);
}

// O = P @ V per (b,h)  (probs f32 -> bf16 on load, Bt = V^T rows)
__global__ void xattn_av_gemm(const float* __restrict__ P,
                              const __bf16* __restrict__ Vt,
                              __bf16* __restrict__ O)
{
  __shared__ __bf16 As[A_BUF];
  __shared__ __bf16 Bs[B_BUF];
  const long bh = blockIdx.z;
  const long b = bh / H_, h = bh % H_;
  gemm_block<false, true, false>(P + bh * (long)NQ_ * NK_, NK_,
                                 Vt + bh * (long)HD_ * NK_, NK_,
                                 O + (b * NQ_) * (long)D_ + h * HD_, D_,
                                 nullptr, 1.0f, NK_,
                                 (long)blockIdx.y * 128, (long)blockIdx.x * 64, As, Bs);
}

// final: out = O @ Wo^T + bo  (TDM ping-pong staging)
__global__ void xattn_out_gemm(const __bf16* __restrict__ O,
                               const __bf16* __restrict__ Wob,
                               const float* __restrict__ bo,
                               float* __restrict__ Out)
{
  __shared__ __bf16 As[2 * A_BUF];
  __shared__ __bf16 Bs[2 * B_BUF];
  gemm_block<true, false, true>(O, D_, Wob, D_, Out, D_, bo, 1.0f, D_,
                                (long)blockIdx.y * 128, (long)blockIdx.x * 64, As, Bs);
}

// ---------------- elementwise kernels ----------------

__global__ void xattn_cvt_bf16(const float* __restrict__ s, __bf16* __restrict__ d, int n)
{
  int i = blockIdx.x * 256 + threadIdx.x;
  if (i < n) d[i] = (__bf16)s[i];
}

// RoPE-xpos + relayout (B,N,D)f32 -> (B,H,N,HD)bf16. One thread per even/odd pair.
__global__ void xattn_rope(const float* __restrict__ Raw,
                           const float* __restrict__ freqs,  // HD/2
                           __bf16* __restrict__ Dst,
                           int Nseq, float inv_scale_base, float sign)
{
  long idx = (long)blockIdx.x * blockDim.x + threadIdx.x;  // (b,n,h,p)
  int  p = idx & 31;  long t = idx >> 5;
  int  h = (int)(t % H_);  t /= H_;
  long n = t % Nseq;  long b = t / Nseq;
  if (b >= B_) return;

  const float* src = Raw + (b * Nseq + n) * (long)D_ + h * HD_ + 2 * p;
  float x1 = src[0], x2 = src[1];

  float ang = freqs[p] * (float)n;
  float s, c; sincosf(ang, &s, &c);
  float power = ((float)n - 0.5f * (float)Nseq) * inv_scale_base;
  float sv = (2.0f * (float)p + 0.4f * (float)HD_) / (1.4f * (float)HD_);
  float sc = powf(sv, sign * power);

  __bf16* dst = Dst + ((b * H_ + h) * (long)Nseq + n) * HD_ + 2 * p;
  dst[0] = (__bf16)((x1 * c - x2 * s) * sc);
  dst[1] = (__bf16)((x2 * c + x1 * s) * sc);
}

// (B,NK,D)f32 -> V^T (B,H,HD,NK)bf16 ; index over output for coalesced stores
__global__ void xattn_vtrans(const float* __restrict__ Vraw, __bf16* __restrict__ Vt)
{
  long idx = (long)blockIdx.x * blockDim.x + threadIdx.x;  // (b,h,e,n)
  long n = idx % NK_;  long t = idx / NK_;
  int  e = (int)(t % HD_);  t /= HD_;
  int  h = (int)(t % H_);   long b = t / H_;
  if (b >= B_) return;
  Vt[idx] = (__bf16)Vraw[(b * NK_ + n) * (long)D_ + h * HD_ + e];
}

// row softmax over 2048-wide rows; one wave32 per row, 8 rows per block
__global__ void xattn_softmax(float* __restrict__ S)
{
  const long row = (long)blockIdx.x * 8 + (threadIdx.x >> 5);
  const int lane = threadIdx.x & 31;
  float* p = S + row * (long)NK_;

  float v[64];
  float mx = -INFINITY;
#pragma unroll
  for (int i = 0; i < 64; ++i) { v[i] = p[lane + i * 32]; mx = fmaxf(mx, v[i]); }
#pragma unroll
  for (int off = 16; off > 0; off >>= 1) mx = fmaxf(mx, __shfl_xor(mx, off, 32));
  float sum = 0.0f;
#pragma unroll
  for (int i = 0; i < 64; ++i) { v[i] = __expf(v[i] - mx); sum += v[i]; }
#pragma unroll
  for (int off = 16; off > 0; off >>= 1) sum += __shfl_xor(sum, off, 32);
  const float inv = 1.0f / sum;
#pragma unroll
  for (int i = 0; i < 64; ++i) p[lane + i * 32] = v[i] * inv;
}

// ---------------- launcher ----------------

extern "C" void kernel_launch(void* const* d_in, const int* in_sizes, int n_in,
                              void* d_out, int out_size, void* d_ws, size_t ws_size,
                              hipStream_t stream)
{
  (void)in_sizes; (void)n_in; (void)out_size; (void)ws_size;

  const float* q_seq  = (const float*)d_in[0];
  const float* kv_seq = (const float*)d_in[1];
  const float* Wq = (const float*)d_in[2];  const float* bq = (const float*)d_in[3];
  const float* Wk = (const float*)d_in[4];  const float* bk = (const float*)d_in[5];
  const float* Wv = (const float*)d_in[6];  const float* bv = (const float*)d_in[7];
  const float* Wo = (const float*)d_in[8];  const float* bo = (const float*)d_in[9];
  const float* fq = (const float*)d_in[10];
  const float* fk = (const float*)d_in[11];

  float* out_main = (float*)d_out;                                  // B*NQ*D
  float* attn     = (float*)d_out + (size_t)B_ * NQ_ * D_;          // B*H*NQ*NK

  // workspace carve-up
  char* ws = (char*)d_ws;
  const size_t WSZ = (size_t)D_ * D_ * sizeof(__bf16);              // 2 MiB
  __bf16* Wqb = (__bf16*)(ws + 0 * WSZ);
  __bf16* Wkb = (__bf16*)(ws + 1 * WSZ);
  __bf16* Wvb = (__bf16*)(ws + 2 * WSZ);
  __bf16* Wob = (__bf16*)(ws + 3 * WSZ);
  float*  raw = (float*) (ws + 4 * WSZ);                            // 16 MiB, reused Q->K->V
  char*   p2  = ws + 4 * WSZ + (size_t)B_ * NQ_ * D_ * sizeof(float);
  const size_t HSZ = (size_t)B_ * NQ_ * D_ * sizeof(__bf16);        // 8 MiB
  __bf16* Qb = (__bf16*)(p2 + 0 * HSZ);   // (B,H,NQ,HD)
  __bf16* Kb = (__bf16*)(p2 + 1 * HSZ);   // (B,H,NK,HD)
  __bf16* Vt = (__bf16*)(p2 + 2 * HSZ);   // (B,H,HD,NK)
  __bf16* Ob = (__bf16*)(p2 + 3 * HSZ);   // (B,NQ,D)

  const int nW = D_ * D_;
  const dim3 blk(256);

  // weights -> bf16
  xattn_cvt_bf16<<<(nW + 255) / 256, blk, 0, stream>>>(Wq, Wqb, nW);
  xattn_cvt_bf16<<<(nW + 255) / 256, blk, 0, stream>>>(Wk, Wkb, nW);
  xattn_cvt_bf16<<<(nW + 255) / 256, blk, 0, stream>>>(Wv, Wvb, nW);
  xattn_cvt_bf16<<<(nW + 255) / 256, blk, 0, stream>>>(Wo, Wob, nW);

  const dim3 gProj(D_ / 64, (B_ * NQ_) / 128);         // 16 x 32
  const int  ropeBlocks = (B_ * NQ_ * H_ * 32) / 256;  // 8192

  // Q path
  xattn_proj_gemm<<<gProj, blk, 0, stream>>>(q_seq, Wqb, bq, raw);
  xattn_rope<<<ropeBlocks, blk, 0, stream>>>(raw, fq, Qb, NQ_, 1.0f / (2.0f * NQ_), +1.0f);
  // K path
  xattn_proj_gemm<<<gProj, blk, 0, stream>>>(kv_seq, Wkb, bk, raw);
  xattn_rope<<<ropeBlocks, blk, 0, stream>>>(raw, fk, Kb, NK_, 1.0f / (2.0f * NK_), -1.0f);
  // V path
  xattn_proj_gemm<<<gProj, blk, 0, stream>>>(kv_seq, Wvb, bv, raw);
  xattn_vtrans<<<(int)(((long)B_ * H_ * HD_ * NK_) / 256), blk, 0, stream>>>(raw, Vt);

  // scores + softmax (attn lives directly in d_out)
  xattn_qk_gemm<<<dim3(NK_ / 64, NQ_ / 128, B_ * H_), blk, 0, stream>>>(Qb, Kb, attn);
  xattn_softmax<<<(B_ * H_ * NQ_) / 8, blk, 0, stream>>>(attn);

  // O = P @ V, then out = O @ Wo^T + bo
  xattn_av_gemm<<<dim3(HD_ / 64, NQ_ / 128, B_ * H_), blk, 0, stream>>>(attn, Vt, Ob);
  xattn_out_gemm<<<dim3(D_ / 64, (B_ * NQ_) / 128), blk, 0, stream>>>(Ob, Wob, bo, out_main);
}
